// MultiHeadAttention_82643760710289
// MI455X (gfx1250) — matrix operand order repaired and checked by reading, not compiled, not run on hardware
//
#include <hip/hip_runtime.h>

// MI455X / gfx1250: wave32, WMMA 16x16x32 bf16 -> f32 accumulate.
// MHA forward: bf16 prepass -> QKV GEMM -> flash attention -> output GEMM.
// B=4, S=2048, H=768, NH=12, HD=64.

typedef __attribute__((ext_vector_type(16))) __bf16 v16bf;
typedef __attribute__((ext_vector_type(8)))  float  v8f;
typedef __attribute__((ext_vector_type(4)))  int    v4i_t;

union F8     { v8f  v; float    f[8]; };
union BF16x16{ v16bf v; unsigned u[8]; };

__device__ __forceinline__ unsigned short f2bf(float f) {
  unsigned u = __builtin_bit_cast(unsigned, f);
  u += 0x7FFFu + ((u >> 16) & 1u);              // round-to-nearest-even
  return (unsigned short)(u >> 16);
}
__device__ __forceinline__ unsigned pack2bf(float lo, float hi) {
  return (unsigned)f2bf(lo) | ((unsigned)f2bf(hi) << 16);
}

// ---- CDNA5 async global->LDS copy (ASYNCcnt path), with sync fallback -----
#if defined(__AMDGCN__) && __has_builtin(__builtin_amdgcn_global_load_async_to_lds_b128) && \
    __has_builtin(__builtin_amdgcn_s_wait_asynccnt)
#define USE_ASYNC_LDS 1
#else
#define USE_ASYNC_LDS 0
#endif

__device__ __forceinline__ void copy16_g2lds(const void* g, void* l) {
#if USE_ASYNC_LDS
  __builtin_amdgcn_global_load_async_to_lds_b128(
      (__attribute__((address_space(1))) v4i_t*)g,
      (__attribute__((address_space(3))) v4i_t*)l, 0, 0);
#else
  *(uint4*)l = *(const uint4*)g;
#endif
}
__device__ __forceinline__ void wait_async_copies() {
#if USE_ASYNC_LDS
  __builtin_amdgcn_s_wait_asynccnt(0);
#endif
}

// A-matrix 16x32 bf16 fragment (ISA 7.12.2): lane m = lane&15,
// VGPR v=g*4+j holds K pair {g*16 + half*8 + 2j, +1}. base = u32 view of tile.
__device__ __forceinline__ v16bf load_a_frag(const unsigned* base, int strideU32, int lane) {
  const int m = lane & 15, h = lane >> 4;
  BF16x16 f;
#pragma unroll
  for (int g = 0; g < 2; ++g)
#pragma unroll
    for (int j = 0; j < 4; ++j)
      f.u[g * 4 + j] = base[m * strideU32 + g * 8 + h * 4 + j];
  return f.v;
}

// B-matrix 32x16 bf16 fragment: lane n = lane&15, VGPR v=g*4+j holds K pair
// {g*16 + half*8 + 2j, +1}. Tile stored column(n)-major with K contiguous.
__device__ __forceinline__ v16bf load_b_frag(const unsigned* base, int strideU32, int lane) {
  const int n = lane & 15, h = lane >> 4;
  BF16x16 f;
#pragma unroll
  for (int v = 0; v < 8; ++v)
    f.u[v] = base[n * strideU32 + (v >> 2) * 8 + h * 4 + (v & 3)];
  return f.v;
}

__device__ __forceinline__ v8f wmma_bf16(v16bf a, v16bf b, v8f c) {
  return __builtin_amdgcn_wmma_f32_16x16x32_bf16(false, a, false, b, (short)0, c, false, false);
}

// ---------------------------------------------------------------------------
// Prepass: f32 -> packed bf16 pairs.
// ---------------------------------------------------------------------------
__global__ __launch_bounds__(256)
void f32_to_bf16_kernel(const float* __restrict__ src, unsigned* __restrict__ dst, int nPairs) {
  int i = blockIdx.x * 256 + threadIdx.x;
  int stride = gridDim.x * 256;
  for (; i < nPairs; i += stride)
    dst[i] = pack2bf(src[2 * i], src[2 * i + 1]);
}

// ---------------------------------------------------------------------------
// Tiled GEMM: Out[M x N] = A[M x 768] * B[768 x N], all-bf16 inputs, f32 acc.
// Workgroup tile 64x128, 8 waves, each wave 32x32 (2x2 WMMA accumulators).
// A tile staged via async b128 copies; B tile transposed to [n][k] in LDS.
// ---------------------------------------------------------------------------
template <bool OUT_BF16>
__global__ __launch_bounds__(256)
void gemm_wmma_kernel(const unsigned short* __restrict__ Abf,
                      const unsigned short* __restrict__ Bbf,
                      void* __restrict__ Op, int N) {
  constexpr int K = 768;
  __shared__ unsigned short aLds[64 * 32];    // [row][k] bf16
  __shared__ unsigned short bLds[128 * 32];   // [n][k]   bf16 (K contiguous)

  const int tid  = threadIdx.x;
  const int wave = tid >> 5, lane = tid & 31;
  const int m0 = blockIdx.x * 64;
  const int n0 = blockIdx.y * 128;
  const int wm = (wave >> 2) * 32;
  const int wn = (wave & 3) * 32;

  F8 acc[2][2];
#pragma unroll
  for (int i = 0; i < 2; ++i)
#pragma unroll
    for (int j = 0; j < 2; ++j)
#pragma unroll
      for (int r = 0; r < 8; ++r) acc[i][j].f[r] = 0.f;

  for (int k0 = 0; k0 < K; k0 += 32) {
    __syncthreads();
    {   // A tile 64x32 bf16: one 16-byte chunk per thread (async -> LDS)
      int row = tid >> 2;
      int c8  = (tid & 3) * 8;
      copy16_g2lds(Abf + (size_t)(m0 + row) * K + k0 + c8, aLds + row * 32 + c8);
    }
    // B tile 32x128 bf16 -> LDS transposed to [n][k]
#pragma unroll
    for (int it = 0; it < 8; ++it) {
      int i = tid + it * 256;                 // 0..2047
      int n2 = i & 63, kk = i >> 6;           // column-pair, k row
      unsigned w2 = *(const unsigned*)(Bbf + (size_t)(k0 + kk) * N + n0 + 2 * n2);
      bLds[(2 * n2) * 32 + kk]     = (unsigned short)(w2 & 0xFFFFu);
      bLds[(2 * n2 + 1) * 32 + kk] = (unsigned short)(w2 >> 16);
    }
    wait_async_copies();
    __syncthreads();

    v16bf afr[2], bfr[2];
#pragma unroll
    for (int mi = 0; mi < 2; ++mi)
      afr[mi] = load_a_frag((const unsigned*)aLds + (wm + mi * 16) * 16, 16, lane);
#pragma unroll
    for (int ni = 0; ni < 2; ++ni)
      bfr[ni] = load_b_frag((const unsigned*)bLds + (wn + ni * 16) * 16, 16, lane);
#pragma unroll
    for (int mi = 0; mi < 2; ++mi)
#pragma unroll
      for (int ni = 0; ni < 2; ++ni)
        acc[mi][ni].v = wmma_bf16(afr[mi], bfr[ni], acc[mi][ni].v);
  }

  // Epilogue: C layout -> element (hl*8+r, lane&15) per 16x16 tile.
  const int nIdx = lane & 15, hl = lane >> 4;
#pragma unroll
  for (int mi = 0; mi < 2; ++mi)
#pragma unroll
    for (int ni = 0; ni < 2; ++ni) {
      int col = n0 + wn + ni * 16 + nIdx;
#pragma unroll
      for (int r = 0; r < 8; ++r) {
        int row = m0 + wm + mi * 16 + hl * 8 + r;
        if (OUT_BF16)
          ((unsigned short*)Op)[(size_t)row * N + col] = f2bf(acc[mi][ni].f[r]);
        else
          ((float*)Op)[(size_t)row * N + col] = acc[mi][ni].f[r];
      }
    }
}

// ---------------------------------------------------------------------------
// Flash attention. qkv (bf16) layout: [b*S + s][2304] with Q|K|V blocks of 768
// and head h at offset h*64.  Grid (S/64, NH, B); 128 threads = 4 waves; each
// wave owns 16 q-rows.  Online softmax in f32, score/PV MACs in bf16 WMMA.
// ---------------------------------------------------------------------------
__global__ __launch_bounds__(128)
void flash_attn_kernel(const unsigned short* __restrict__ qkv, unsigned short* __restrict__ ctx) {
  constexpr int   S = 2048;
  constexpr float SCALE = 0.125f;               // 1/sqrt(64)
  __shared__ unsigned short kLds[64 * 64];      // [kpos][d], d contiguous
  __shared__ unsigned short vLds[64 * 64];      // [d][kpos], kpos contiguous
  __shared__ unsigned short pLds[4 * 16 * 64];  // per-wave P tiles [16][64]

  const int tid  = threadIdx.x;
  const int wave = tid >> 5, lane = tid & 31;
  const int q0 = blockIdx.x * 64;
  const int h  = blockIdx.y;
  const int b  = blockIdx.z;
  const int qBase = q0 + wave * 16;
  const int nIdx = lane & 15, hl = lane >> 4;
  const unsigned* qkvU = (const unsigned*)qkv;

  // Q fragments (d chunks 0..31, 32..63), kept in registers for entire loop.
  const unsigned* qsrc = qkvU + (size_t)(b * S + qBase) * 1152 + h * 32;
  v16bf aQ[2];
  aQ[0] = load_a_frag(qsrc,      1152, lane);
  aQ[1] = load_a_frag(qsrc + 16, 1152, lane);

  F8 o[4];
  float mrow[8], lrow[8];
#pragma unroll
  for (int r = 0; r < 8; ++r) { mrow[r] = -1e30f; lrow[r] = 0.f; }
#pragma unroll
  for (int dt = 0; dt < 4; ++dt)
#pragma unroll
    for (int r = 0; r < 8; ++r) o[dt].f[r] = 0.f;

  const int kBlocks = blockIdx.x + 1;           // causal: only k <= q0+63
  for (int kb = 0; kb < kBlocks; ++kb) {
    const int kStart = kb * 64;
    __syncthreads();
    {   // K tile via async b128 copies (row-major, matches memory)
      const unsigned short* ksrc = qkv + (size_t)(b * S + kStart) * 2304 + 768 + h * 64;
#pragma unroll
      for (int it = 0; it < 4; ++it) {
        int i = tid + it * 128;                 // 0..511
        int srow = i >> 3, c8 = (i & 7) * 8;
        copy16_g2lds(ksrc + (size_t)srow * 2304 + c8, kLds + srow * 64 + c8);
      }
      // V tile transposed to [d][kpos] (async copy cannot transpose)
      const unsigned* vsrc = qkvU + (size_t)(b * S + kStart) * 1152 + 768 + h * 32;
#pragma unroll
      for (int it = 0; it < 16; ++it) {
        int i = tid + it * 128;
        int srow = i >> 5, c2 = i & 31;         // kpos, d-pair
        unsigned vv = vsrc[(size_t)srow * 1152 + c2];
        vLds[(2 * c2)     * 64 + srow] = (unsigned short)(vv & 0xFFFFu);
        vLds[(2 * c2 + 1) * 64 + srow] = (unsigned short)(vv >> 16);
      }
    }
    wait_async_copies();
    __syncthreads();

    const bool diag = (kb == kBlocks - 1);
#pragma unroll
    for (int t = 0; t < 4; ++t) {               // 16 k-positions per tile
      F8 s;
#pragma unroll
      for (int r = 0; r < 8; ++r) s.f[r] = 0.f;
      s.v = wmma_bf16(aQ[0], load_b_frag((const unsigned*)kLds + t * 512,      32, lane), s.v);
      s.v = wmma_bf16(aQ[1], load_b_frag((const unsigned*)kLds + t * 512 + 16, 32, lane), s.v);

#pragma unroll
      for (int r = 0; r < 8; ++r) {
        float sv = s.f[r] * SCALE;
        if (diag) {
          int q  = qBase + hl * 8 + r;
          int kp = kStart + t * 16 + nIdx;
          if (kp > q) sv = -1e30f;
        }
        s.f[r] = sv;
      }
      // online softmax, row-wise across the 16-lane half groups
#pragma unroll
      for (int r = 0; r < 8; ++r) {
        float x = s.f[r];
        float rmax = x;
        rmax = fmaxf(rmax, __shfl_xor(rmax, 1));
        rmax = fmaxf(rmax, __shfl_xor(rmax, 2));
        rmax = fmaxf(rmax, __shfl_xor(rmax, 4));
        rmax = fmaxf(rmax, __shfl_xor(rmax, 8));
        float mNew  = fmaxf(mrow[r], rmax);
        float alpha = __expf(mrow[r] - mNew);
        float p     = __expf(x - mNew);
        mrow[r] = mNew;
        lrow[r] = lrow[r] * alpha + p;          // per-lane partial row-sum
#pragma unroll
        for (int dt = 0; dt < 4; ++dt) o[dt].f[r] *= alpha;
        pLds[wave * 1024 + (hl * 8 + r) * 64 + t * 16 + nIdx] = f2bf(p);
      }
    }

    // P @ V : reload P as A fragments (same-wave LDS is in-order).
    v16bf aP0 = load_a_frag((const unsigned*)pLds + wave * 512,      32, lane);
    v16bf aP1 = load_a_frag((const unsigned*)pLds + wave * 512 + 16, 32, lane);
#pragma unroll
    for (int dt = 0; dt < 4; ++dt) {
      o[dt].v = wmma_bf16(aP0, load_b_frag((const unsigned*)vLds + dt * 512,      32, lane), o[dt].v);
      o[dt].v = wmma_bf16(aP1, load_b_frag((const unsigned*)vLds + dt * 512 + 16, 32, lane), o[dt].v);
    }
  }

  // normalize and write context (bf16) as [b*S+s][768] with head offset h*64
  float inv[8];
#pragma unroll
  for (int r = 0; r < 8; ++r) {
    float l = lrow[r];
    l += __shfl_xor(l, 1); l += __shfl_xor(l, 2);
    l += __shfl_xor(l, 4); l += __shfl_xor(l, 8);
    inv[r] = 1.0f / l;
  }
#pragma unroll
  for (int dt = 0; dt < 4; ++dt)
#pragma unroll
    for (int r = 0; r < 8; ++r) {
      int row = b * S + qBase + hl * 8 + r;
      ctx[(size_t)row * 768 + h * 64 + dt * 16 + nIdx] = f2bf(o[dt].f[r] * inv[r]);
    }
}

// ---------------------------------------------------------------------------
extern "C" void kernel_launch(void* const* d_in, const int* in_sizes, int n_in,
                              void* d_out, int out_size, void* d_ws, size_t ws_size,
                              hipStream_t stream) {
  const float* x     = (const float*)d_in[0];   // [4,2048,768]
  const float* w_qkv = (const float*)d_in[1];   // [768,2304]
  const float* w_out = (const float*)d_in[2];   // [768,768]
  float* out = (float*)d_out;                   // [4,2048,768]

  // workspace layout (bf16 elements)
  unsigned short* qkv_bf  = (unsigned short*)d_ws;                  // 8192*2304
  unsigned short* ctx_bf  = qkv_bf  + (size_t)8192 * 2304;          // 8192*768
  unsigned short* x_bf    = ctx_bf  + (size_t)8192 * 768;           // 8192*768
  unsigned short* wqkv_bf = x_bf    + (size_t)8192 * 768;           // 768*2304
  unsigned short* wout_bf = wqkv_bf + (size_t)768 * 2304;           // 768*768

  // 0) one-time f32 -> bf16 conversion (keeps GEMM inner loops conversion-free)
  int px = 8192 * 768 / 2, pq = 768 * 2304 / 2, po = 768 * 768 / 2;
  f32_to_bf16_kernel<<<(px + 255) / 256, 256, 0, stream>>>(x,     (unsigned*)x_bf,    px);
  f32_to_bf16_kernel<<<(pq + 255) / 256, 256, 0, stream>>>(w_qkv, (unsigned*)wqkv_bf, pq);
  f32_to_bf16_kernel<<<(po + 255) / 256, 256, 0, stream>>>(w_out, (unsigned*)wout_bf, po);

  // 1) QKV projection
  dim3 g1(8192 / 64, 2304 / 128);
  gemm_wmma_kernel<true><<<g1, 256, 0, stream>>>(x_bf, wqkv_bf, qkv_bf, 2304);

  // 2) causal flash attention over 48 (b,h) pairs
  dim3 g2(2048 / 64, 12, 4);
  flash_attn_kernel<<<g2, 128, 0, stream>>>(qkv_bf, ctx_bf);

  // 3) output projection -> f32
  dim3 g3(8192 / 64, 768 / 128);
  gemm_wmma_kernel<false><<<g3, 256, 0, stream>>>(ctx_bf, wout_bf, out, 768);
}
